// Ernie4_5_MoeMLP_9904194585277
// MI455X (gfx1250) — compile-verified, matrix operand order checked by reading, not run on hardware
//
#include <hip/hip_runtime.h>
#include <hip/hip_bf16.h>

// ---------------- problem constants ----------------
#define S_TOK 16384
#define H_DIM 1024
#define I_DIM 512
#define E_EXP 64
#define TOPK  2
#define CAPE  640

// ---------------- WMMA types ----------------
typedef __bf16 v16bf __attribute__((ext_vector_type(16)));
typedef float  v8f   __attribute__((ext_vector_type(8)));

union Frag {
    v16bf v;
    uint4 q[2];
};

__device__ __forceinline__ unsigned short f2bf_bits(float f) {
    unsigned int u = __float_as_uint(f);
    unsigned int lsb = (u >> 16) & 1u;
    u += 0x7fffu + lsb;
    return (unsigned short)(u >> 16);
}
__device__ __forceinline__ float bf2f(unsigned short b) {
    return __uint_as_float(((unsigned int)b) << 16);
}

// ---------------- async global -> LDS copy (CDNA5) ----------------
#if __has_builtin(__builtin_amdgcn_global_load_async_to_lds_b128) && \
    __has_builtin(__builtin_amdgcn_s_wait_asynccnt)
#define HAVE_ASYNC_LDS 1
#else
#define HAVE_ASYNC_LDS 0
#endif

typedef int v4i __attribute__((ext_vector_type(4)));
typedef __attribute__((address_space(1))) v4i GV4;   // global int4
typedef __attribute__((address_space(3))) v4i LV4;   // LDS int4

__device__ __forceinline__ void cp16_g2l(const unsigned short* g, unsigned short* l) {
#if HAVE_ASYNC_LDS
    // emits global_load_async_to_lds_b128 (tracked by ASYNCcnt)
    __builtin_amdgcn_global_load_async_to_lds_b128((GV4*)g, (LV4*)l, 0, 0);
#else
    *(uint4*)l = *(const uint4*)g;
#endif
}
__device__ __forceinline__ void async_wait_all() {
#if HAVE_ASYNC_LDS
    __builtin_amdgcn_s_wait_asynccnt(0);
#endif
}

// =====================================================================
// Kernel A: gate logits + softmax + top-2 (one wave32 per token)
// =====================================================================
__global__ __launch_bounds__(128)
void gate_topk_kernel(const float* __restrict__ x, const float* __restrict__ wgate,
                      float* __restrict__ logits_out,
                      float* __restrict__ topk_p, int* __restrict__ topk_e)
{
    const int wave = threadIdx.x >> 5;
    const int lane = threadIdx.x & 31;
    const int s = blockIdx.x * 4 + wave;
    const float* xs = x + (long long)s * H_DIM;

    float a0 = 0.f, a1 = 0.f;
    #pragma unroll 4
    for (int h = 0; h < H_DIM; ++h) {
        float xv = xs[h];
        a0 = fmaf(xv, wgate[h * E_EXP + lane], a0);
        a1 = fmaf(xv, wgate[h * E_EXP + lane + 32], a1);
    }
    logits_out[(long long)s * E_EXP + lane]      = a0;
    logits_out[(long long)s * E_EXP + lane + 32] = a1;

    // softmax over 64 experts (2 per lane)
    float m = fmaxf(a0, a1);
    for (int off = 16; off; off >>= 1) m = fmaxf(m, __shfl_xor(m, off, 32));
    float p0 = __expf(a0 - m), p1 = __expf(a1 - m);
    float sum = p0 + p1;
    for (int off = 16; off; off >>= 1) sum += __shfl_xor(sum, off, 32);
    p0 /= sum; p1 /= sum;

    // local sorted pair (tie -> lower index; lane < lane+32 always)
    float c0p, c1p; int c0e, c1e;
    if (p0 >= p1) { c0p = p0; c0e = lane;      c1p = p1; c1e = lane + 32; }
    else          { c0p = p1; c0e = lane + 32; c1p = p0; c1e = lane;      }

    // merge sorted pairs across the wave
    for (int off = 16; off; off >>= 1) {
        float o0p = __shfl_xor(c0p, off, 32); int o0e = __shfl_xor(c0e, off, 32);
        float o1p = __shfl_xor(c1p, off, 32); int o1e = __shfl_xor(c1e, off, 32);
        bool firstMine = (c0p > o0p) || (c0p == o0p && c0e < o0e);
        float n0p, n1p; int n0e, n1e;
        if (firstMine) {
            n0p = c0p; n0e = c0e;
            bool secMine = (c1p > o0p) || (c1p == o0p && c1e < o0e);
            if (secMine) { n1p = c1p; n1e = c1e; } else { n1p = o0p; n1e = o0e; }
        } else {
            n0p = o0p; n0e = o0e;
            bool secOther = (o1p > c0p) || (o1p == c0p && o1e < c0e);
            if (secOther) { n1p = o1p; n1e = o1e; } else { n1p = c0p; n1e = c0e; }
        }
        c0p = n0p; c0e = n0e; c1p = n1p; c1e = n1e;
    }
    if (lane == 0) {
        topk_p[s * 2] = c0p; topk_p[s * 2 + 1] = c1p;
        topk_e[s * 2] = c0e; topk_e[s * 2 + 1] = c1e;
    }
}

// =====================================================================
// Kernel B: deterministic token-major capacity assignment.
// One wave32 per expert scans all routes; ballot prefix gives the slot.
// =====================================================================
__global__ __launch_bounds__(32)
void slots_kernel(const int* __restrict__ topk_e, int* __restrict__ slots)
{
    const int e = blockIdx.x;
    const int lane = threadIdx.x;
    int base = 0;
    for (int i = 0; i < S_TOK * TOPK; i += 32) {
        int re = topk_e[i + lane];
        bool match = (re == e);
        unsigned mask = (unsigned)__ballot(match);
        if (match) {
            int pos = base + __popc(mask & ((1u << lane) - 1u));
            slots[i + lane] = (pos < CAPE) ? (e * CAPE + pos) : -1;
        }
        base += __popc(mask);
    }
}

// =====================================================================
// Kernel C: normalized combine weights + router loss
// =====================================================================
__global__ __launch_bounds__(256)
void cw_kernel(const int* __restrict__ slots, const float* __restrict__ topk_p,
               float* __restrict__ cw_out, float* __restrict__ loss_out)
{
    int s = blockIdx.x * blockDim.x + threadIdx.x;
    if (s == 0) loss_out[0] = 0.f;
    if (s >= S_TOK) return;
    float w0 = (slots[s * 2]     >= 0) ? topk_p[s * 2]     : 0.f;
    float w1 = (slots[s * 2 + 1] >= 0) ? topk_p[s * 2 + 1] : 0.f;
    float d = w0 + w1;
    d = (d < 1e-12f) ? 1e-12f : d;
    cw_out[s * 2]     = w0 / d;
    cw_out[s * 2 + 1] = w1 / d;
}

// =====================================================================
// Kernel D: per-expert transpose fp32 -> bf16 (so GEMM B operands are
// stored N-major, K-minor and fragments load as contiguous 16B chunks)
// =====================================================================
__global__ __launch_bounds__(256)
void transpose_bf16_kernel(const float* __restrict__ src, unsigned short* __restrict__ dst,
                           int rows, int cols)
{
    __shared__ float tile[32][33];
    const int e = blockIdx.z;
    const float* s = src + (long long)e * rows * cols;
    unsigned short* d = dst + (long long)e * rows * cols;
    const int c0 = blockIdx.x * 32, r0 = blockIdx.y * 32;
    for (int y = threadIdx.y; y < 32; y += 8)
        tile[y][threadIdx.x] = s[(long long)(r0 + y) * cols + c0 + threadIdx.x];
    __syncthreads();
    for (int y = threadIdx.y; y < 32; y += 8)
        d[(long long)(c0 + y) * rows + r0 + threadIdx.x] = f2bf_bits(tile[threadIdx.x][y]);
}

// =====================================================================
// Kernel E0/E1: zero dispatch buffer, scatter tokens into [E,CAP,H] bf16
// =====================================================================
__global__ __launch_bounds__(256)
void zero_u4_kernel(uint4* __restrict__ p, long long n4)
{
    uint4 z = {0u, 0u, 0u, 0u};
    for (long long i = blockIdx.x * 256LL + threadIdx.x; i < n4; i += (long long)gridDim.x * 256LL)
        p[i] = z;
}

__global__ __launch_bounds__(256)
void scatter_kernel(const float* __restrict__ x, const int* __restrict__ slots,
                    unsigned short* __restrict__ Xd)
{
    const int r = blockIdx.x;          // route index [0, S*K)
    const int slot = slots[r];
    if (slot < 0) return;
    const float* xs = x + (long long)(r / TOPK) * H_DIM;
    unsigned short* dstp = Xd + (long long)slot * H_DIM;
    for (int c = threadIdx.x; c < H_DIM; c += 256)
        dstp[c] = f2bf_bits(xs[c]);
}

// =====================================================================
// WMMA grouped GEMM:  C[e] = A[e] (MxK, bf16 row-major) @ B[e]^T
// Bt stored [N,K] bf16 row-major. Block tile 64x64, BK=32, 4 waves
// (2x2), wave tile 32x32 -> 2x2 v_wmma_f32_16x16x32_bf16 per k-step.
// Double-buffered LDS staged via GLOBAL_LOAD_ASYNC_TO_LDS_B128
// (ASYNCcnt) when available; sync fallback otherwise.
// FUSED: two B matrices (gate/up), epilogue h = silu(g)*u -> bf16.
// =====================================================================
#define BM 64
#define BN 64
#define BK 32
#define LDP 40   // padded LDS row stride in bf16 units (80 bytes)

template <bool FUSED>
__global__ __launch_bounds__(128)
void gemm_wmma_kernel(const unsigned short* __restrict__ A, long long strideA, int lda,
                      const unsigned short* __restrict__ B1, long long strideB, int ldb,
                      const unsigned short* __restrict__ B2,
                      unsigned short* __restrict__ Out, long long strideO, int ldo,
                      int Kdim)
{
    __shared__ unsigned short sA [2][BM * LDP];
    __shared__ unsigned short sB1[2][BN * LDP];
    __shared__ unsigned short sB2[FUSED ? 2 : 1][FUSED ? (BN * LDP) : 16];

    const int e = blockIdx.z;
    const unsigned short* Ae  = A  + (long long)e * strideA;
    const unsigned short* B1e = B1 + (long long)e * strideB;
    const unsigned short* B2e = FUSED ? (B2 + (long long)e * strideB) : B1;
    unsigned short* Oe = Out + (long long)e * strideO;

    const int m0 = blockIdx.y * BM;
    const int n0 = blockIdx.x * BN;

    const int tid  = threadIdx.x;
    const int lane = tid & 31;
    const int w    = tid >> 5;       // 0..3
    const int wm   = w >> 1;         // 0..1
    const int wn   = w & 1;          // 0..1
    const int lr   = lane & 15;      // row/col within 16
    const int lh   = lane >> 4;      // half select

    // staging role: each thread moves 32B of each tile row-half
    const int row_ld = tid >> 1;          // 0..63
    const int ch0    = (tid & 1) * 16;    // bf16 offset: 0 or 16

    v8f acc1[2][2];
    v8f acc2[2][2];
    #pragma unroll
    for (int i = 0; i < 2; ++i)
        #pragma unroll
        for (int j = 0; j < 2; ++j) { acc1[i][j] = (v8f)0.0f; acc2[i][j] = (v8f)0.0f; }

    auto stage = [&](int buf, int k0) {
        const unsigned short* ga = Ae + (long long)(m0 + row_ld) * lda + k0 + ch0;
        cp16_g2l(ga,     &sA[buf][row_ld * LDP + ch0]);
        cp16_g2l(ga + 8, &sA[buf][row_ld * LDP + ch0 + 8]);
        const unsigned short* gb = B1e + (long long)(n0 + row_ld) * ldb + k0 + ch0;
        cp16_g2l(gb,     &sB1[buf][row_ld * LDP + ch0]);
        cp16_g2l(gb + 8, &sB1[buf][row_ld * LDP + ch0 + 8]);
        if (FUSED) {
            const unsigned short* gu = B2e + (long long)(n0 + row_ld) * ldb + k0 + ch0;
            cp16_g2l(gu,     &sB2[buf][row_ld * LDP + ch0]);
            cp16_g2l(gu + 8, &sB2[buf][row_ld * LDP + ch0 + 8]);
        }
    };

    stage(0, 0);   // prologue: fill buffer 0

    for (int k0 = 0; k0 < Kdim; k0 += BK) {
        const int buf = (k0 / BK) & 1;

        async_wait_all();      // this wave's async copies into `buf` landed
        __syncthreads();       // whole workgroup's copies visible; prev compute done

        if (k0 + BK < Kdim)
            stage(buf ^ 1, k0 + BK);   // overlap next tile load with compute

        // ---- load fragments (A: row M, K pairs in dwords; B from [N,K]) ----
        Frag a[2], b1[2], b2[2];
        #pragma unroll
        for (int mt = 0; mt < 2; ++mt) {
            const int row = wm * 32 + mt * 16 + lr;
            a[mt].q[0] = *(const uint4*)(&sA[buf][row * LDP +      lh * 8]);
            a[mt].q[1] = *(const uint4*)(&sA[buf][row * LDP + 16 + lh * 8]);
        }
        #pragma unroll
        for (int nt = 0; nt < 2; ++nt) {
            const int row = wn * 32 + nt * 16 + lr;
            b1[nt].q[0] = *(const uint4*)(&sB1[buf][row * LDP +      lh * 8]);
            b1[nt].q[1] = *(const uint4*)(&sB1[buf][row * LDP + 16 + lh * 8]);
            if (FUSED) {
                b2[nt].q[0] = *(const uint4*)(&sB2[buf][row * LDP +      lh * 8]);
                b2[nt].q[1] = *(const uint4*)(&sB2[buf][row * LDP + 16 + lh * 8]);
            }
        }
        #pragma unroll
        for (int mt = 0; mt < 2; ++mt)
            #pragma unroll
            for (int nt = 0; nt < 2; ++nt) {
                acc1[mt][nt] = __builtin_amdgcn_wmma_f32_16x16x32_bf16(
                    false, a[mt].v, false, b1[nt].v, (short)0, acc1[mt][nt], false, false);
                if (FUSED)
                    acc2[mt][nt] = __builtin_amdgcn_wmma_f32_16x16x32_bf16(
                        false, a[mt].v, false, b2[nt].v, (short)0, acc2[mt][nt], false, false);
            }
    }

    // ---- epilogue ----
    #pragma unroll
    for (int mt = 0; mt < 2; ++mt)
        #pragma unroll
        for (int nt = 0; nt < 2; ++nt)
            #pragma unroll
            for (int j = 0; j < 8; ++j) {
                const int row = m0 + wm * 32 + mt * 16 + lh * 8 + j;
                const int col = n0 + wn * 32 + nt * 16 + lr;
                float v = acc1[mt][nt][j];
                if (FUSED) {
                    float g = v, u = acc2[mt][nt][j];
                    v = g / (1.f + __expf(-g)) * u;   // silu(g) * u
                }
                Oe[(long long)row * ldo + col] = f2bf_bits(v);
            }
}

// =====================================================================
// Kernel H: combine  out[s] = sum_k cw[s,k] * expert_out[slot[s,k]]
// =====================================================================
__global__ __launch_bounds__(256)
void combine_kernel(const unsigned short* __restrict__ eo, const int* __restrict__ slots,
                    const float* __restrict__ cw, float* __restrict__ out)
{
    const int s = blockIdx.x;
    const int s0 = slots[s * 2], s1 = slots[s * 2 + 1];
    const float w0 = cw[s * 2], w1 = cw[s * 2 + 1];
    for (int c = threadIdx.x; c < H_DIM; c += 256) {
        float acc = 0.f;
        if (s0 >= 0) acc += w0 * bf2f(eo[(long long)s0 * H_DIM + c]);
        if (s1 >= 0) acc += w1 * bf2f(eo[(long long)s1 * H_DIM + c]);
        out[(long long)s * H_DIM + c] = acc;
    }
}

// =====================================================================
// Host launcher
// =====================================================================
extern "C" void kernel_launch(void* const* d_in, const int* in_sizes, int n_in,
                              void* d_out, int out_size, void* d_ws, size_t ws_size,
                              hipStream_t stream)
{
    (void)in_sizes; (void)n_in; (void)out_size; (void)ws_size;

    const float* x      = (const float*)d_in[0];   // [S,H]
    const float* w_gate = (const float*)d_in[1];   // [H,E]
    const float* wg     = (const float*)d_in[2];   // [E,H,I]
    const float* wu     = (const float*)d_in[3];   // [E,H,I]
    const float* wd     = (const float*)d_in[4];   // [E,I,H]

    // ---- output layout: out | cw | router_loss | logits ----
    float* o_out    = (float*)d_out;
    float* o_cw     = o_out + (size_t)S_TOK * H_DIM;
    float* o_loss   = o_cw + (size_t)S_TOK * TOPK;
    float* o_logits = o_loss + 1;

    // ---- workspace layout ----
    char* ws = (char*)d_ws;
    size_t off = 0;
    auto take = [&](size_t bytes) { char* p = ws + off; off += (bytes + 255) & ~(size_t)255; return p; };

    float*          topk_p = (float*)take((size_t)S_TOK * TOPK * 4);
    int*            topk_e = (int*)  take((size_t)S_TOK * TOPK * 4);
    int*            slots  = (int*)  take((size_t)S_TOK * TOPK * 4);
    unsigned short* WgT    = (unsigned short*)take((size_t)E_EXP * I_DIM * H_DIM * 2);
    unsigned short* WuT    = (unsigned short*)take((size_t)E_EXP * I_DIM * H_DIM * 2);
    unsigned short* WdT    = (unsigned short*)take((size_t)E_EXP * H_DIM * I_DIM * 2);
    unsigned short* Xd     = (unsigned short*)take((size_t)E_EXP * CAPE * H_DIM * 2);
    unsigned short* Hbuf   = (unsigned short*)take((size_t)E_EXP * CAPE * I_DIM * 2);
    unsigned short* EO     = (unsigned short*)take((size_t)E_EXP * CAPE * H_DIM * 2);

    // 1. gate + softmax + top2 (+ logits output)
    gate_topk_kernel<<<S_TOK / 4, 128, 0, stream>>>(x, w_gate, o_logits, topk_p, topk_e);

    // 2. deterministic capacity slots
    slots_kernel<<<E_EXP, 32, 0, stream>>>(topk_e, slots);

    // 3. normalized combine weights + loss
    cw_kernel<<<(S_TOK + 255) / 256, 256, 0, stream>>>(slots, topk_p, o_cw, o_loss);

    // 4. weights -> bf16 transposed ([N,K] for the WMMA B operand)
    transpose_bf16_kernel<<<dim3(I_DIM / 32, H_DIM / 32, E_EXP), dim3(32, 8), 0, stream>>>(
        wg, WgT, H_DIM, I_DIM);
    transpose_bf16_kernel<<<dim3(I_DIM / 32, H_DIM / 32, E_EXP), dim3(32, 8), 0, stream>>>(
        wu, WuT, H_DIM, I_DIM);
    transpose_bf16_kernel<<<dim3(H_DIM / 32, I_DIM / 32, E_EXP), dim3(32, 8), 0, stream>>>(
        wd, WdT, I_DIM, H_DIM);

    // 5. dispatch x -> bf16 [E,CAP,H] (zero-fill then scatter)
    {
        long long n4 = ((long long)E_EXP * CAPE * H_DIM * 2) / 16;
        zero_u4_kernel<<<4096, 256, 0, stream>>>((uint4*)Xd, n4);
        scatter_kernel<<<S_TOK * TOPK, 256, 0, stream>>>(x, slots, Xd);
    }

    // 6. GEMM1 fused: h = silu(Xd @ Wg) * (Xd @ Wu)   [E, CAP, I] bf16
    gemm_wmma_kernel<true><<<dim3(I_DIM / BN, CAPE / BM, E_EXP), 128, 0, stream>>>(
        Xd,  (long long)CAPE * H_DIM, H_DIM,
        WgT, (long long)I_DIM * H_DIM, H_DIM,
        WuT,
        Hbuf, (long long)CAPE * I_DIM, I_DIM,
        H_DIM);

    // 7. GEMM2: eo = h @ Wd   [E, CAP, H] bf16
    gemm_wmma_kernel<false><<<dim3(H_DIM / BN, CAPE / BM, E_EXP), 128, 0, stream>>>(
        Hbuf, (long long)CAPE * I_DIM, I_DIM,
        WdT,  (long long)H_DIM * I_DIM, I_DIM,
        nullptr,
        EO,   (long long)CAPE * H_DIM, H_DIM,
        I_DIM);

    // 8. combine
    combine_kernel<<<S_TOK, 256, 0, stream>>>(EO, slots, o_cw, o_out);
}
